// LSTMModel_48249662603421
// MI455X (gfx1250) — matrix operand order repaired and checked by reading, not compiled, not run on hardware
//
#include <hip/hip_runtime.h>
#include <hip/hip_bf16.h>

// ---------------------------------------------------------------------------
// 2-layer LSTM (B=4096, T=512, H1=H2=512) + softmax head for MI455X (gfx1250).
// bf16 WMMA (v_wmma_f32_16x16x32_bf16), fp32 accumulation; cell state, biases,
// rank-1 x-term and classifier head stay fp32.  Weights pre-transposed
// (WT[n][k]) so both WMMA fragments are contiguous ds_load_b128.  LDS tiles
// are double-buffered and staged with async global->LDS copies (ASYNCcnt)
// when the toolchain exposes the builtin.
// ---------------------------------------------------------------------------

typedef __attribute__((ext_vector_type(8)))  __bf16 v8bf;
typedef __attribute__((ext_vector_type(16))) __bf16 v16bf;
typedef __attribute__((ext_vector_type(8)))  float  v8f;
typedef __attribute__((ext_vector_type(4)))  int    v4i;

#define LSTM_B   4096
#define LSTM_T   512
#define LSTM_H   512
#define FOURH    2048
#define NUM_CLS  20
#define LDS_RS   40            // padded LDS row stride (bf16): 80B, bank-conflict free

#define AS1 __attribute__((address_space(1)))
#define AS3 __attribute__((address_space(3)))

#ifndef __has_builtin
#define __has_builtin(x) 0
#endif

#if __has_builtin(__builtin_amdgcn_global_load_async_to_lds_b128)
#define ASYNC_LDS 1
#else
#define ASYNC_LDS 0
#endif

__device__ __forceinline__ float sigmoidf_(float x) {
    return 1.0f / (1.0f + __expf(-x));
}

// Stage 16 bytes (8 bf16) global -> LDS.
__device__ __forceinline__ void stage16(const __bf16* __restrict__ g, __bf16* l) {
#if ASYNC_LDS
    __builtin_amdgcn_global_load_async_to_lds_b128(
        (AS1 v4i*)(v4i*)g, (AS3 v4i*)(v4i*)l, 0, 0);
#else
    *reinterpret_cast<uint4*>(l) = *reinterpret_cast<const uint4*>(g);
#endif
}

__device__ __forceinline__ void wait_async_copies() {
#if ASYNC_LDS
#if __has_builtin(__builtin_amdgcn_s_wait_asynccnt)
    __builtin_amdgcn_s_wait_asynccnt(0);
#else
    asm volatile("s_wait_asynccnt 0x0" ::: "memory");
#endif
#endif
}

// ------------------------- utility kernels ---------------------------------

__global__ __launch_bounds__(256) void zero16_kernel(uint4* __restrict__ p, long n16) {
    long i = (long)blockIdx.x * 256 + threadIdx.x;
    if (i < n16) p[i] = make_uint4(0u, 0u, 0u, 0u);
}

// dst[n*K + k] = (bf16) src[k*2048 + n]   (transpose + fp32->bf16, one-time)
__global__ __launch_bounds__(256)
void f2bf_transpose_kernel(const float* __restrict__ src, __bf16* __restrict__ dst, int K) {
    long i = (long)blockIdx.x * 256 + threadIdx.x;           // over 2048*K
    if (i >= (long)FOURH * K) return;
    const int n = (int)(i / K);
    const int k = (int)(i % K);
    dst[(size_t)n * K + k] = (__bf16)src[(size_t)k * FOURH + n];
}

// ------------------------- fused LSTM cell GEMM ----------------------------
//
//   z = [A] @ W + bias (+ x_t * w0 for layer 1),  A = Aa (K=512) or [Aa|Ab] (K=1024)
//   i,j,f,o = split(z);  c = c*sig(f+1) + sig(i)*tanh(j);  h = tanh(c)*sig(o)
//
// Block: 256 threads = 8 waves.  Tile: 128 batch rows x 32 h-cols x 4 gates.
// Wave (wm,wn) in 4x2: 32 rows x 16 cols, 2 M-subtiles x 4 gates = 8 WMMAs/chunk.
// Double-buffered LDS; chunk i+1 prefetched (async) while chunk i computes.

__global__ __launch_bounds__(256)
void lstm_cell_kernel(const __bf16* __restrict__ Aa,   // [B, 512] bf16 (h input)
                      const __bf16* __restrict__ Ab,   // [B, 512] bf16 or null
                      const __bf16* __restrict__ WT,   // [2048, KT] bf16 (transposed W)
                      const float*  __restrict__ bias, // [2048]
                      const float*  __restrict__ x,    // [B, T] fp32 or null
                      const float*  __restrict__ w0,   // [2048] fp32 (W1 row 0) or null
                      int t,
                      float*  __restrict__ C,          // [B, 512] in/out
                      __bf16* __restrict__ Hb,         // [B, 512] out bf16
                      float*  __restrict__ Hf)         // [B, 512] out fp32 or null
{
    __shared__ __bf16 ldsA[2][128 * LDS_RS];      // A tile: [row][k], padded rows
    __shared__ __bf16 ldsW[2][4 * 32 * LDS_RS];   // WT tile: [gate][n][k], padded rows

    const int tid  = threadIdx.x;
    const int lane = tid & 31;
    const int wv   = tid >> 5;
    const int wm   = wv >> 1;               // 0..3 : M sub-block (32 rows)
    const int wn   = wv & 1;                // 0..1 : N sub-block (16 cols)
    const int m0   = blockIdx.x * 128;      // 32 blocks
    const int n0   = blockIdx.y * 32;       // 16 blocks

    const int mloc = lane & 15;
    const int hi   = lane >> 4;

    const int nParts  = (Ab != nullptr) ? 2 : 1;
    const int KT      = nParts * LSTM_H;    // WT row length
    const int nChunks = nParts * (LSTM_H / 32);

    // ---- staging thread mapping (each thread stages 16 bf16 of A and W) ----
    const int rA   = tid >> 1;              // 0..127 : A row
    const int segA = tid & 1;               // k 0..15 / 16..31
    const int gW   = tid >> 6;              // 0..3   : gate
    const int remW = tid & 63;
    const int nnW  = remW >> 1;             // 0..31  : WT row in tile
    const int segW = remW & 1;

    __bf16* dstA[2];
    __bf16* dstW[2];
#pragma unroll
    for (int bfr = 0; bfr < 2; ++bfr) {
        dstA[bfr] = &ldsA[bfr][rA * LDS_RS + segA * 16];
        dstW[bfr] = &ldsW[bfr][(gW * 32 + nnW) * LDS_RS + segW * 16];
    }
    const size_t  aRowOff  = (size_t)(m0 + rA) * LSTM_H + segA * 16;
    const __bf16* wRowBase = WT + (size_t)(gW * LSTM_H + n0 + nnW) * KT + segW * 16;

    auto prefetch = [&](int ci, int bfr) {
        const int part = ci >> 4;                     // 16 chunks per 512-K part
        const int kc   = (ci & 15) * 32;
        const __bf16* Ap = part ? Ab : Aa;
        const __bf16* ga = Ap + aRowOff + kc;
        stage16(ga,     dstA[bfr]);
        stage16(ga + 8, dstA[bfr] + 8);
        const __bf16* gw = wRowBase + (size_t)ci * 32;    // part*512 + kc == ci*32
        stage16(gw,     dstW[bfr]);
        stage16(gw + 8, dstW[bfr] + 8);
    };

    v8f acc[4][2];
#pragma unroll
    for (int g = 0; g < 4; ++g)
#pragma unroll
        for (int s = 0; s < 2; ++s)
#pragma unroll
            for (int e = 0; e < 8; ++e) acc[g][s][e] = 0.0f;

    prefetch(0, 0);

    for (int ci = 0; ci < nChunks; ++ci) {
        const int bfr = ci & 1;
        // Our async copies for chunk ci are outstanding; wait + block-sync.
        // (A wave only reaches this barrier after its WMMAs consumed the other
        //  buffer, so issuing the next prefetch after it is hazard-free.)
        wait_async_copies();
        __syncthreads();
        if (ci + 1 < nChunks) prefetch(ci + 1, bfr ^ 1);

        // ---- A fragments: 16-bit A 16x32 layout (lane half hi holds
        //      K = hi*8..+7 in elems 0..7 and K = 16+hi*8..+7 in 8..15) ----
        v16bf a0, a1;
        {
            const __bf16* ar0 = &ldsA[bfr][(wm * 32 + mloc) * LDS_RS];
            const __bf16* ar1 = &ldsA[bfr][(wm * 32 + 16 + mloc) * LDS_RS];
            union { v16bf v; v8bf h[2]; } u0, u1;
            u0.h[0] = *reinterpret_cast<const v8bf*>(ar0 + hi * 8);
            u0.h[1] = *reinterpret_cast<const v8bf*>(ar0 + 16 + hi * 8);
            u1.h[0] = *reinterpret_cast<const v8bf*>(ar1 + hi * 8);
            u1.h[1] = *reinterpret_cast<const v8bf*>(ar1 + 16 + hi * 8);
            a0 = u0.v;
            a1 = u1.v;
        }

        // ---- B fragments + WMMA per gate (B 32x16: lane half hi holds
        //      K = hi*8..+7 in elems 0..7 and K = 16+hi*8..+7 in 8..15, column = lane&15) ----
        const int nloc = wn * 16 + (lane & 15);
        const int kb   = hi * 8;
#pragma unroll
        for (int g = 0; g < 4; ++g) {
            const __bf16* br = &ldsW[bfr][(g * 32 + nloc) * LDS_RS + kb];
            union { v16bf v; v8bf h[2]; } ub;
            ub.h[0] = *reinterpret_cast<const v8bf*>(br);
            ub.h[1] = *reinterpret_cast<const v8bf*>(br + 16);
            acc[g][0] = __builtin_amdgcn_wmma_f32_16x16x32_bf16(
                false, a0, false, ub.v, (short)0, acc[g][0], false, false);
            acc[g][1] = __builtin_amdgcn_wmma_f32_16x16x32_bf16(
                false, a1, false, ub.v, (short)0, acc[g][1], false, false);
        }
    }

    // ---------------- epilogue: fused LSTM cell update ----------------
    // C/D layout: lane l -> col = l&15, row = (l>>4)*8 + r (per 16x16 tile).
    const int col = n0 + wn * 16 + (lane & 15);
    const float bi = bias[col];
    const float bj = bias[LSTM_H + col];
    const float bf = bias[2 * LSTM_H + col];
    const float bo = bias[3 * LSTM_H + col];
    float wi = 0.f, wj = 0.f, wf = 0.f, wo = 0.f;
    if (w0 != nullptr) {
        wi = w0[col];
        wj = w0[LSTM_H + col];
        wf = w0[2 * LSTM_H + col];
        wo = w0[3 * LSTM_H + col];
    }

#pragma unroll
    for (int s = 0; s < 2; ++s) {
#pragma unroll
        for (int r = 0; r < 8; ++r) {
            const int row = m0 + wm * 32 + s * 16 + hi * 8 + r;
            float zi = acc[0][s][r] + bi;
            float zj = acc[1][s][r] + bj;
            float zf = acc[2][s][r] + bf;
            float zo = acc[3][s][r] + bo;
            if (x != nullptr) {
                const float xv = x[(size_t)row * LSTM_T + t];
                zi += xv * wi; zj += xv * wj; zf += xv * wf; zo += xv * wo;
            }
            const size_t idx = (size_t)row * LSTM_H + col;
            const float c_old = C[idx];
            const float c_new = c_old * sigmoidf_(zf + 1.0f)   // forget bias 1.0
                              + sigmoidf_(zi) * tanhf(zj);
            const float h_new = tanhf(c_new) * sigmoidf_(zo);
            C[idx]  = c_new;
            Hb[idx] = (__bf16)h_new;
            if (Hf != nullptr) Hf[idx] = h_new;
        }
    }
}

// ------------------------- classifier head ---------------------------------

__global__ __launch_bounds__(256)
void head_kernel(const float* __restrict__ h2,  // [B, 512] fp32
                 const float* __restrict__ sw,  // [512, 20]
                 const float* __restrict__ sb,  // [20]
                 float* __restrict__ out)       // logits | probs | preds(int)
{
    const int b = blockIdx.x * 256 + threadIdx.x;
    if (b >= LSTM_B) return;

    float acc[NUM_CLS];
#pragma unroll
    for (int c = 0; c < NUM_CLS; ++c) acc[c] = sb[c];

    const float* hr = h2 + (size_t)b * LSTM_H;
    for (int k = 0; k < LSTM_H; ++k) {
        const float hv = hr[k];
        const float* wr = sw + (size_t)k * NUM_CLS;
#pragma unroll
        for (int c = 0; c < NUM_CLS; ++c) acc[c] += hv * wr[c];
    }

    float mx = acc[0];
    int am = 0;
#pragma unroll
    for (int c = 1; c < NUM_CLS; ++c)
        if (acc[c] > mx) { mx = acc[c]; am = c; }  // first max kept (argmax semantics)

    float e[NUM_CLS], se = 0.0f;
#pragma unroll
    for (int c = 0; c < NUM_CLS; ++c) { e[c] = __expf(acc[c] - mx); se += e[c]; }
    const float inv = 1.0f / se;

    float* logits = out;
    float* probs  = out + (size_t)LSTM_B * NUM_CLS;
    int*   preds  = (int*)(out + (size_t)2 * LSTM_B * NUM_CLS);
#pragma unroll
    for (int c = 0; c < NUM_CLS; ++c) {
        logits[(size_t)b * NUM_CLS + c] = acc[c];
        probs [(size_t)b * NUM_CLS + c] = e[c] * inv;
    }
    preds[b] = am;
}

// ------------------------------ launcher -----------------------------------

extern "C" void kernel_launch(void* const* d_in, const int* in_sizes, int n_in,
                              void* d_out, int out_size, void* d_ws, size_t ws_size,
                              hipStream_t stream) {
    const float* x  = (const float*)d_in[0];   // [4096, 512]
    const float* W1 = (const float*)d_in[1];   // [513, 2048]
    const float* b1 = (const float*)d_in[2];   // [2048]
    const float* W2 = (const float*)d_in[3];   // [1024, 2048]
    const float* b2 = (const float*)d_in[4];   // [2048]
    const float* sw = (const float*)d_in[5];   // [512, 20]
    const float* sb = (const float*)d_in[6];   // [20]
    float* out = (float*)d_out;

    char* ws = (char*)d_ws;
    const size_t SZF = (size_t)LSTM_B * LSTM_H * sizeof(float);   // 8 MB
    const size_t SZB = (size_t)LSTM_B * LSTM_H * sizeof(__bf16);  // 4 MB

    float*  c1      = (float*)(ws);
    float*  c2      = (float*)(ws + SZF);
    __bf16* h1b[2]  = { (__bf16*)(ws + 2 * SZF),
                        (__bf16*)(ws + 2 * SZF + SZB) };
    __bf16* h2b[2]  = { (__bf16*)(ws + 2 * SZF + 2 * SZB),
                        (__bf16*)(ws + 2 * SZF + 3 * SZB) };
    float*  h2f     = (float*)(ws + 2 * SZF + 4 * SZB);
    __bf16* WT1     = (__bf16*)(ws + 3 * SZF + 4 * SZB);                              // [2048][512]
    __bf16* WT2     = (__bf16*)(ws + 3 * SZF + 4 * SZB + (size_t)FOURH * LSTM_H * 2); // [2048][1024]

    // Zero c1, c2 and both h ping-pong buffer sets (32 MB).
    {
        const long n16 = (long)((2 * SZF + 4 * SZB) / 16);
        zero16_kernel<<<dim3((unsigned)((n16 + 255) / 256)), dim3(256), 0, stream>>>(
            (uint4*)ws, n16);
    }
    // One-time: transpose + convert weights to bf16.  WT[n][k] = W[k][n].
    // W1 rows 1..512 only; row 0 (scalar-input weights) stays fp32 and is
    // applied exactly in the epilogue.
    {
        const long n1 = (long)FOURH * LSTM_H;
        f2bf_transpose_kernel<<<dim3((unsigned)((n1 + 255) / 256)), dim3(256), 0, stream>>>(
            W1 + FOURH, WT1, LSTM_H);
        const long n2 = (long)FOURH * (2 * LSTM_H);
        f2bf_transpose_kernel<<<dim3((unsigned)((n2 + 255) / 256)), dim3(256), 0, stream>>>(
            W2, WT2, 2 * LSTM_H);
    }

    const dim3 grid(LSTM_B / 128, LSTM_H / 32);  // 32 x 16 = 512 blocks
    const dim3 block(256);

    for (int t = 0; t < LSTM_T; ++t) {
        const int cur = t & 1;
        // Layer 1: z = h1_prev @ W1[1:,:] + x_t * W1[0,:] + b1
        lstm_cell_kernel<<<grid, block, 0, stream>>>(
            h1b[cur], nullptr, WT1, b1, x, W1 /* row 0 */, t,
            c1, h1b[cur ^ 1], nullptr);
        // Layer 2: z = [h1_new | h2_prev] @ W2 + b2
        lstm_cell_kernel<<<grid, block, 0, stream>>>(
            h1b[cur ^ 1], h2b[cur], WT2, b2, nullptr, nullptr, t,
            c2, h2b[cur ^ 1], h2f);
    }

    head_kernel<<<dim3(LSTM_B / 256), dim3(256), 0, stream>>>(h2f, sw, sb, out);
}
